// SynthesizerAttention_65034394796821
// MI455X (gfx1250) — compile-verified, matrix operand order checked by reading, not run on hardware
//
#include <hip/hip_runtime.h>

#define NHEAD 16
#define HD 64
#define TSEQ 2048
#define CDIM 1024
#define BATCH 2
#define NEGV -1e10f

typedef __attribute__((ext_vector_type(16))) __bf16 v16bf;
typedef __attribute__((ext_vector_type(8)))  float  v8f;

#define WMMA_BF16(A, B, C) \
  __builtin_amdgcn_wmma_f32_16x16x32_bf16(false, (A), false, (B), (short)0, (C), false, false)

static __device__ __forceinline__ unsigned short f2bf(float f) {
  union { float f; unsigned int u; } v; v.f = f;
  unsigned int u = v.u;
  unsigned int r = u + 0x7FFFu + ((u >> 16) & 1u);   // round-to-nearest-even
  return (unsigned short)(r >> 16);
}

// Load a 16x32 bf16 fragment (A layout) / 32x16 (B layout, rows = weight rows).
// lane 0-15: row (row0+lane), k=[k0,k0+16); lane 16-31: row (row0+lane-16), k=[k0+16,k0+32)
static __device__ __forceinline__ v16bf load_frag(const unsigned short* base, int row0,
                                                  int ld, int k0, int lane) {
  const int r = lane & 15, h = lane >> 4;
  return *reinterpret_cast<const v16bf*>(base + (size_t)(row0 + r) * ld + k0 + h * 16);
}

// ---------------------------------------------------------------- conversions
// 4 elements per thread: float4 in -> packed 2x(2xbf16) out
__global__ void cvt_bf16_v4(const float4* __restrict__ src, uint2* __restrict__ dst, int n4) {
  int i = blockIdx.x * blockDim.x + threadIdx.x;
  if (i < n4) {
    float4 f = src[i];
    uint2 o;
    o.x = (unsigned int)f2bf(f.x) | ((unsigned int)f2bf(f.y) << 16);
    o.y = (unsigned int)f2bf(f.z) | ((unsigned int)f2bf(f.w) << 16);
    dst[i] = o;
  }
}

__global__ void transpose_w2(const float* __restrict__ w2, unsigned short* __restrict__ w2t) {
  // w2: (HD, 2048) row-major -> w2t: (TSEQ, HD)
  int i = blockIdx.x * blockDim.x + threadIdx.x;
  if (i < HD * TSEQ) {
    int d = i / TSEQ, s = i % TSEQ;
    w2t[s * HD + d] = f2bf(w2[i]);
  }
}

// ------------------------------------------------- k = relu(x@w1^T+b1), v = x@wv^T+bv
// kb: (B,H,T,HD) bf16 ; vt: (B,H,HD,T) bf16 (transposed for P@V B-operand)
// Software-pipelined and unrolled x2 so the double buffers ping-pong (no reg moves).
__global__ __launch_bounds__(128) void gemm_kv(
    const unsigned short* __restrict__ xb,   // (B*T, C) bf16
    const unsigned short* __restrict__ w1b,  // (C, C) bf16, row = out dim
    const unsigned short* __restrict__ wvb,  // (C, C) bf16
    const float* __restrict__ b1, const float* __restrict__ bv,
    unsigned short* __restrict__ kb,
    unsigned short* __restrict__ vt) {
  const int lane = threadIdx.x & 31;
  const int wave = threadIdx.x >> 5;
  const int m0 = blockIdx.x * 64 + wave * 16;
  const int n0 = blockIdx.y * 64;

  v8f accK[4] = {}, accV[4] = {};

  // prologue: load fragments for kk = 0
  v16bf a = load_frag(xb, m0, CDIM, 0, lane);
  v16bf bK[4], bV[4];
#pragma unroll
  for (int nt = 0; nt < 4; ++nt) {
    bK[nt] = load_frag(w1b, n0 + nt * 16, CDIM, 0, lane);
    bV[nt] = load_frag(wvb, n0 + nt * 16, CDIM, 0, lane);
  }

#pragma unroll 2
  for (int kk = 0; kk < CDIM; kk += 32) {
    // issue next iteration's loads first (kn wraps harmlessly on last iter)
    const int kn = (kk + 32 < CDIM) ? kk + 32 : 0;
    v16bf an = load_frag(xb, m0, CDIM, kn, lane);
    __builtin_prefetch(xb + (size_t)(m0 + (lane & 15)) * CDIM + kn, 0, 1);
    v16bf bKn[4], bVn[4];
#pragma unroll
    for (int nt = 0; nt < 4; ++nt) {
      bKn[nt] = load_frag(w1b, n0 + nt * 16, CDIM, kn, lane);
      bVn[nt] = load_frag(wvb, n0 + nt * 16, CDIM, kn, lane);
    }
    // consume current fragments (loaded one iteration ago -> latency hidden)
#pragma unroll
    for (int nt = 0; nt < 4; ++nt) {
      accK[nt] = WMMA_BF16(a, bK[nt], accK[nt]);
      accV[nt] = WMMA_BF16(a, bV[nt], accV[nt]);
    }
    a = an;
#pragma unroll
    for (int nt = 0; nt < 4; ++nt) { bK[nt] = bKn[nt]; bV[nt] = bVn[nt]; }
  }

  const int half = lane >> 4, col = lane & 15;
#pragma unroll
  for (int nt = 0; nt < 4; ++nt) {
    const int n = n0 + nt * 16 + col;
    const int h = n >> 6, d = n & 63;
    const float biasK = b1[n], biasV = bv[n];
    // kb scatter: 16-bit stores (t-stride 128B, unpackable)
#pragma unroll
    for (int r = 0; r < 8; ++r) {
      const int m = m0 + r + 8 * half;
      const int b = m >> 11, t = m & (TSEQ - 1);
      float kval = accK[nt][r] + biasK;
      kval = kval > 0.f ? kval : 0.f;
      kb[(((size_t)(b * NHEAD + h)) * TSEQ + t) * HD + d] = f2bf(kval);
    }
    // vt scatter: adjacent r -> adjacent t -> pack 2x bf16 into one b32 store
#pragma unroll
    for (int r = 0; r < 8; r += 2) {
      const int m = m0 + r + 8 * half;
      const int b = m >> 11, t = m & (TSEQ - 1);          // t is even
      const unsigned int lo = f2bf(accV[nt][r] + biasV);
      const unsigned int hi = f2bf(accV[nt][r + 1] + biasV);
      *reinterpret_cast<unsigned int*>(
          vt + (((size_t)(b * NHEAD + h)) * HD + d) * TSEQ + t) = lo | (hi << 16);
    }
  }
}

// ------------------------- fused synthesizer attention (flash-style, no att matrix)
// Per-wave LDS regions are private and LDS executes in order per wave, so no
// block barriers are needed: waves run fully decoupled.
__global__ __launch_bounds__(128) void flash_syn(
    const unsigned short* __restrict__ kb,   // (B,H,T,HD) bf16 (relu'd)
    const unsigned short* __restrict__ w2t,  // (T, HD) bf16
    const float* __restrict__ b2,            // (2048)
    const unsigned short* __restrict__ vt,   // (B,H,HD,T) bf16
    unsigned short* __restrict__ ybuf) {     // (B,T,C) bf16
  __shared__ unsigned short plds[4 * 16 * 64];   // per-wave P tile (16x64 bf16)
  const int lane = threadIdx.x & 31;
  const int wave = threadIdx.x >> 5;
  const int mt = blockIdx.x & 31;
  const int h  = (blockIdx.x >> 5) & 15;
  const int b  = blockIdx.x >> 9;
  const int mb = mt * 64;
  const int trow0 = mb + wave * 16;
  const int half = lane >> 4, col = lane & 15;
  const size_t bh = (size_t)b * NHEAD + h;

  const unsigned short* kbase = kb + bh * TSEQ * HD;
  const unsigned short* vbase = vt + bh * HD * TSEQ;
  unsigned short* myp = plds + wave * (16 * 64);

  // A operand (relu_k rows for this wave's 16 queries), K = hd = 64 -> 2 k-steps
  v16bf a0 = load_frag(kbase, trow0, HD, 0, lane);
  v16bf a1 = load_frag(kbase, trow0, HD, 32, lane);

  v8f yacc[4] = {};
  float rowmax[8], rowsum[8];
#pragma unroll
  for (int r = 0; r < 8; ++r) { rowmax[r] = -3.0e38f; rowsum[r] = 0.f; }

  for (int sb = 0; sb <= mb; sb += 64) {
    // ---- batched score-operand loads (one wait covers all eight)
    v16bf w0[4], w1f[4];
#pragma unroll
    for (int nt = 0; nt < 4; ++nt) {
      w0[nt]  = load_frag(w2t, sb + nt * 16, HD, 0, lane);
      w1f[nt] = load_frag(w2t, sb + nt * 16, HD, 32, lane);
    }
    // ---- warm caches for the next s-block (no register cost)
    if (sb + 64 <= mb) {
      __builtin_prefetch(w2t + (size_t)(sb + 64 + (lane & 15) * 4) * HD, 0, 1);
      __builtin_prefetch(vbase + (size_t)(lane & 15) * 4 * TSEQ + sb + 64, 0, 1);
    }
    // ---- scores S(16x64) = relu_k @ w2t-rows
    v8f sacc[4] = {};
#pragma unroll
    for (int nt = 0; nt < 4; ++nt) {
      sacc[nt] = WMMA_BF16(a0, w0[nt], sacc[nt]);
      sacc[nt] = WMMA_BF16(a1, w1f[nt], sacc[nt]);
    }
    // ---- issue V fragment loads NOW; the softmax VALU work below hides them
    v16bf vb0[4], vb1[4];
#pragma unroll
    for (int dt = 0; dt < 4; ++dt) {
      vb0[dt] = load_frag(vbase, dt * 16, TSEQ, sb, lane);
      vb1[dt] = load_frag(vbase, dt * 16, TSEQ, sb + 32, lane);
    }
    // ---- bias + causal mask
#pragma unroll
    for (int nt = 0; nt < 4; ++nt) {
      const int s = sb + nt * 16 + col;
      const float bias = b2[s];
#pragma unroll
      for (int r = 0; r < 8; ++r) {
        const int t = trow0 + r + 8 * half;
        const float vsc = sacc[nt][r] + bias;
        sacc[nt][r] = (s > t) ? NEGV : vsc;
      }
    }
    // ---- online softmax update; P -> LDS (bf16)
    float newmax[8];
#pragma unroll
    for (int r = 0; r < 8; ++r) {
      float mv = fmaxf(fmaxf(sacc[0][r], sacc[1][r]), fmaxf(sacc[2][r], sacc[3][r]));
      mv = fmaxf(mv, __shfl_xor(mv, 1, 32));
      mv = fmaxf(mv, __shfl_xor(mv, 2, 32));
      mv = fmaxf(mv, __shfl_xor(mv, 4, 32));
      mv = fmaxf(mv, __shfl_xor(mv, 8, 32));
      newmax[r] = fmaxf(rowmax[r], mv);
    }
#pragma unroll
    for (int r = 0; r < 8; ++r) {
      const float scale = __expf(rowmax[r] - newmax[r]);
      float bsum = 0.f;
#pragma unroll
      for (int nt = 0; nt < 4; ++nt) {
        const float p = __expf(sacc[nt][r] - newmax[r]);
        bsum += p;
        myp[(r + 8 * half) * 64 + nt * 16 + col] = f2bf(p);
      }
      bsum += __shfl_xor(bsum, 1, 32);
      bsum += __shfl_xor(bsum, 2, 32);
      bsum += __shfl_xor(bsum, 4, 32);
      bsum += __shfl_xor(bsum, 8, 32);
      rowsum[r] = rowsum[r] * scale + bsum;
      rowmax[r] = newmax[r];
#pragma unroll
      for (int dt = 0; dt < 4; ++dt) yacc[dt][r] *= scale;
    }

    // ---- y += P(16x64) @ V(64x64); per-wave LDS in-order => no barrier needed
    v16bf pa0 = load_frag(myp, 0, 64, 0, lane);
    v16bf pa1 = load_frag(myp, 0, 64, 32, lane);
#pragma unroll
    for (int dt = 0; dt < 4; ++dt) {
      yacc[dt] = WMMA_BF16(pa0, vb0[dt], yacc[dt]);
      yacc[dt] = WMMA_BF16(pa1, vb1[dt], yacc[dt]);
    }
  }

  // ---- normalize (8 reciprocals, not 32 divides) and scatter y -> (B,T,C) bf16
  float inv[8];
#pragma unroll
  for (int r = 0; r < 8; ++r) inv[r] = 1.0f / rowsum[r];
#pragma unroll
  for (int dt = 0; dt < 4; ++dt) {
#pragma unroll
    for (int r = 0; r < 8; ++r) {
      const int t = trow0 + r + 8 * half;
      const int c = h * HD + dt * 16 + col;
      ybuf[((size_t)b * TSEQ + t) * CDIM + c] = f2bf(yacc[dt][r] * inv[r]);
    }
  }
}

// ------------------------------------------------------- out = y @ wp^T + bp (fp32)
__global__ __launch_bounds__(128) void gemm_out(
    const unsigned short* __restrict__ yb,   // (B*T, C) bf16
    const unsigned short* __restrict__ wpb,  // (C, C) bf16
    const float* __restrict__ bp,
    float* __restrict__ out) {
  const int lane = threadIdx.x & 31;
  const int wave = threadIdx.x >> 5;
  const int m0 = blockIdx.x * 64 + wave * 16;
  const int n0 = blockIdx.y * 64;

  v8f acc[4] = {};
  v16bf a = load_frag(yb, m0, CDIM, 0, lane);
  v16bf bw[4];
#pragma unroll
  for (int nt = 0; nt < 4; ++nt) bw[nt] = load_frag(wpb, n0 + nt * 16, CDIM, 0, lane);

#pragma unroll 2
  for (int kk = 0; kk < CDIM; kk += 32) {
    const int kn = (kk + 32 < CDIM) ? kk + 32 : 0;
    v16bf an = load_frag(yb, m0, CDIM, kn, lane);
    v16bf bwn[4];
#pragma unroll
    for (int nt = 0; nt < 4; ++nt) bwn[nt] = load_frag(wpb, n0 + nt * 16, CDIM, kn, lane);
#pragma unroll
    for (int nt = 0; nt < 4; ++nt) acc[nt] = WMMA_BF16(a, bw[nt], acc[nt]);
    a = an;
#pragma unroll
    for (int nt = 0; nt < 4; ++nt) bw[nt] = bwn[nt];
  }

  const int half = lane >> 4, col = lane & 15;
#pragma unroll
  for (int nt = 0; nt < 4; ++nt) {
    const int n = n0 + nt * 16 + col;
    const float bias = bp[n];
#pragma unroll
    for (int r = 0; r < 8; ++r) {
      const int m = m0 + r + 8 * half;
      out[(size_t)m * CDIM + n] = acc[nt][r] + bias;
    }
  }
}

// ------------------------------------------------------------------- launcher
extern "C" void kernel_launch(void* const* d_in, const int* in_sizes, int n_in,
                              void* d_out, int out_size, void* d_ws, size_t ws_size,
                              hipStream_t stream) {
  const float* x  = (const float*)d_in[0];
  const float* w1 = (const float*)d_in[1];
  const float* b1 = (const float*)d_in[2];
  const float* w2 = (const float*)d_in[3];
  const float* b2 = (const float*)d_in[4];
  const float* wv = (const float*)d_in[5];
  const float* bv = (const float*)d_in[6];
  const float* wp = (const float*)d_in[7];
  const float* bp = (const float*)d_in[8];
  float* out = (float*)d_out;
  (void)in_sizes; (void)n_in; (void)out_size; (void)ws_size;

  char* ws = (char*)d_ws;
  size_t off = 0;
  auto carve = [&](size_t bytes) -> char* {
    char* p = ws + off;
    off += (bytes + 255) & ~(size_t)255;
    return p;
  };
  unsigned short* xb  = (unsigned short*)carve((size_t)BATCH * TSEQ * CDIM * 2);
  unsigned short* w1b = (unsigned short*)carve((size_t)CDIM * CDIM * 2);
  unsigned short* wvb = (unsigned short*)carve((size_t)CDIM * CDIM * 2);
  unsigned short* wpb = (unsigned short*)carve((size_t)CDIM * CDIM * 2);
  unsigned short* w2t = (unsigned short*)carve((size_t)TSEQ * HD * 2);
  unsigned short* kb  = (unsigned short*)carve((size_t)BATCH * NHEAD * TSEQ * HD * 2);
  unsigned short* vt  = (unsigned short*)carve((size_t)BATCH * NHEAD * HD * TSEQ * 2);
  unsigned short* yb  = (unsigned short*)carve((size_t)BATCH * TSEQ * CDIM * 2);

  int n4 = (BATCH * TSEQ * CDIM) / 4;
  cvt_bf16_v4<<<(n4 + 255) / 256, 256, 0, stream>>>((const float4*)x, (uint2*)xb, n4);
  n4 = (CDIM * CDIM) / 4;
  cvt_bf16_v4<<<(n4 + 255) / 256, 256, 0, stream>>>((const float4*)w1, (uint2*)w1b, n4);
  cvt_bf16_v4<<<(n4 + 255) / 256, 256, 0, stream>>>((const float4*)wv, (uint2*)wvb, n4);
  cvt_bf16_v4<<<(n4 + 255) / 256, 256, 0, stream>>>((const float4*)wp, (uint2*)wpb, n4);
  int n = HD * TSEQ;
  transpose_w2<<<(n + 255) / 256, 256, 0, stream>>>(w2, w2t);

  dim3 gGemm(BATCH * TSEQ / 64, CDIM / 64);          // 64 x 16 blocks, 128 thr
  gemm_kv<<<gGemm, 128, 0, stream>>>(xb, w1b, wvb, b1, bv, kb, vt);

  flash_syn<<<BATCH * NHEAD * (TSEQ / 64), 128, 0, stream>>>(kb, w2t, b2, vt, yb);

  gemm_out<<<gGemm, 128, 0, stream>>>(yb, wpb, bp, out);
}
